// M_11433202942186
// MI455X (gfx1250) — matrix operand order repaired and checked by reading, not compiled, not run on hardware
//
#include <hip/hip_runtime.h>

typedef _Float16 v8h  __attribute__((ext_vector_type(8)));
typedef _Float16 v16h __attribute__((ext_vector_type(16)));
typedef float    v8f  __attribute__((ext_vector_type(8)));

#define ROWS_PER_BLOCK 128          // rows of x per workgroup (8 waves x 16)
#define KDIM 128                    // LayerNorm / GEMM inner dim
#define NOUT 512                    // output features
#define LDS_STRIDE 136              // halves per padded row (68 dwords -> conflict-free)

// gfx1250 hardware tanh transcendental (branch-free GELU epilogue)
__device__ __forceinline__ float hw_tanhf(float x) {
#if __has_builtin(__builtin_amdgcn_tanhf)
    return __builtin_amdgcn_tanhf(x);
#elif __has_builtin(__builtin_amdgcn_tanh_f32)
    return __builtin_amdgcn_tanh_f32(x);
#else
    float r;
    // v_nop satisfies the TRANS-op result hazard (1 op before use)
    asm volatile("v_tanh_f32 %0, %1\n\tv_nop" : "=v"(r) : "v"(x));
    return r;
#endif
}

__device__ __forceinline__ float gelu_tanh(float y) {
    // 0.5*y*(1+tanh(sqrt(2/pi)*(y + 0.044715*y^3))); max abs err ~3e-4 vs exact
    const float K0  = 0.7978845608028654f;           // sqrt(2/pi)
    const float K01 = 0.7978845608028654f * 0.044715f;
    float y2 = y * y;
    float u  = y * __builtin_fmaf(y2, K01, K0);
    float th = hw_tanhf(u);
    float hy = 0.5f * y;
    return __builtin_fmaf(hy, th, hy);
}

__device__ __forceinline__ void load_btile(const _Float16* __restrict__ wrow,
                                           int kOffB, v16h b[4]) {
    #pragma unroll
    for (int kc = 0; kc < 4; ++kc) {
        v8h lo = *(const v8h*)(wrow + kc * 32 + kOffB);      // K = kb..kb+7
        v8h hi = *(const v8h*)(wrow + kc * 32 + kOffB + 16); // K = kb+16..kb+23
        v16h t;
        #pragma unroll
        for (int e = 0; e < 8; ++e) { t[e] = lo[e]; t[8 + e] = hi[e]; }
        b[kc] = t;
    }
}

extern "C" __global__ __launch_bounds__(256)
void ln_gemm_gelu_kernel(const float* __restrict__ x,
                         const float* __restrict__ gamma,
                         const float* __restrict__ beta,
                         const float* __restrict__ W,      // [512,128] row-major (out,in)
                         const float* __restrict__ bias,   // [512]
                         float* __restrict__ out)          // [rows,512]
{
    extern __shared__ __align__(16) _Float16 smem[];
    _Float16* sA = smem;                                   // 128 * 136 halves
    _Float16* sW = smem + ROWS_PER_BLOCK * LDS_STRIDE;     // 512 * 136 halves

    const int tid  = threadIdx.x;
    const int lane = tid & 31;
    const int wave = tid >> 5;
    const long long block_row = (long long)blockIdx.x * ROWS_PER_BLOCK;

    // ---------------- Stage W: fp32 global -> f16 LDS (padded rows) ----------------
    {
        const float4* W4 = (const float4*)W;               // 16384 float4s
        for (int i = tid; i < (NOUT * KDIM) / 4; i += 256) {
            float4 w = W4[i];
            int base = i << 2;
            int c = base >> 7;                             // output feature
            int k = base & 127;                            // input feature
            _Float16* dst = sW + c * LDS_STRIDE + k;
            dst[0] = (_Float16)w.x; dst[1] = (_Float16)w.y;
            dst[2] = (_Float16)w.z; dst[3] = (_Float16)w.w;
        }
    }

    // ---------------- LayerNorm (fp32 stats) -> f16 LDS A ----------------
    {
        const int row  = tid >> 1;                         // 0..127
        const int half = tid & 1;                          // 64 elems each
        const float4* x4 = (const float4*)(x + (block_row + row) * KDIM + half * 64);
        float4 v[16];
        float s = 0.f, ss = 0.f;
        #pragma unroll
        for (int i = 0; i < 16; ++i) {
            v[i] = x4[i];
            s  += v[i].x + v[i].y + v[i].z + v[i].w;
            ss += v[i].x * v[i].x + v[i].y * v[i].y + v[i].z * v[i].z + v[i].w * v[i].w;
        }
        // partner lane is adjacent (pairs never cross a wave32 boundary)
        s  += __shfl_xor(s, 1, 32);
        ss += __shfl_xor(ss, 1, 32);
        const float mean = s * (1.0f / 128.0f);
        const float var  = ss * (1.0f / 128.0f) - mean * mean;
        const float inv  = rsqrtf(var + 1e-6f);

        const float4* g4 = (const float4*)(gamma + half * 64);
        const float4* b4 = (const float4*)(beta  + half * 64);
        _Float16* dst = sA + row * LDS_STRIDE + half * 64;
        #pragma unroll
        for (int i = 0; i < 16; ++i) {
            float4 g = g4[i], b = b4[i];
            dst[4*i+0] = (_Float16)((v[i].x - mean) * inv * g.x + b.x);
            dst[4*i+1] = (_Float16)((v[i].y - mean) * inv * g.y + b.y);
            dst[4*i+2] = (_Float16)((v[i].z - mean) * inv * g.z + b.z);
            dst[4*i+3] = (_Float16)((v[i].w - mean) * inv * g.w + b.w);
        }
    }
    __syncthreads();

    // ---------------- A fragments (ISA 7.12.2 16-bit A 16x32 layout) ----------------
    // lanes 0-15: M = lane,    K = {kb..kb+7, kb+16..kb+23}, kb = 32*kc
    // lanes 16-31: M = lane-16, same but kb += 8
    v16h afrag[4];
    {
        const int m    = lane & 15;
        const int koff = (lane >> 4) ? 8 : 0;
        const _Float16* arow = sA + (wave * 16 + m) * LDS_STRIDE;
        #pragma unroll
        for (int kc = 0; kc < 4; ++kc) {
            const int kb = kc * 32 + koff;
            v8h lo = *(const v8h*)(arow + kb);        // K = kb .. kb+7
            v8h hi = *(const v8h*)(arow + kb + 16);   // K = kb+16 .. kb+23
            v16h a;
            #pragma unroll
            for (int e = 0; e < 8; ++e) { a[e] = lo[e]; a[8 + e] = hi[e]; }
            afrag[kc] = a;
        }
    }

    // ---------------- 32 column tiles, double-buffered B, full unroll ----------------
    // B 32x16 f16 layout: lane n (0-15) holds col n, K = {kb..kb+7, kb+16..kb+23} (kb = 32*kc);
    // lanes 16-31 hold same col with kb += 8.
    const int cLane = lane & 15;
    const int kOffB = (lane >> 4) ? 8 : 0;
    const long long out_row0 = block_row + wave * 16 + ((lane >= 16) ? 8 : 0);

    float* __restrict__ obase       = out + out_row0 * (long long)NOUT + cLane; // + j*512 + t*16
    const float* __restrict__ bptr  = bias + cLane;                             // + t*16
    const _Float16* __restrict__ w0 = sW + cLane * LDS_STRIDE;                  // + t*16*LDS_STRIDE

    v16h bfrag[2][4];
    float bc[2];
    load_btile(w0, kOffB, bfrag[0]);
    bc[0] = bptr[0];

    #pragma unroll
    for (int t = 0; t < 32; ++t) {
        const int cur = t & 1, nxt = cur ^ 1;
        // Prefetch next tile's B fragments + bias while this tile's WMMAs run
        if (t + 1 < 32) {
            load_btile(w0 + (t + 1) * 16 * LDS_STRIDE, kOffB, bfrag[nxt]);
            bc[nxt] = bptr[(t + 1) * 16];
        }
        v8f acc = {};
        #pragma unroll
        for (int kc = 0; kc < 4; ++kc) {
            acc = __builtin_amdgcn_wmma_f32_16x16x32_f16(
                /*neg_a=*/false, afrag[kc], /*neg_b=*/false, bfrag[cur][kc],
                /*c_mod=*/(short)0, acc, /*reuse_a=*/false, /*reuse_b=*/false);
        }
        const float bcv = bc[cur];
        #pragma unroll
        for (int j = 0; j < 8; ++j) {
            obase[j * NOUT + t * 16] = gelu_tanh(acc[j] + bcv);
        }
    }
}

extern "C" void kernel_launch(void* const* d_in, const int* in_sizes, int n_in,
                              void* d_out, int out_size, void* d_ws, size_t ws_size,
                              hipStream_t stream) {
    const float* x     = (const float*)d_in[0];   // 64*56*56*128
    const float* gamma = (const float*)d_in[1];   // 128
    const float* beta  = (const float*)d_in[2];   // 128
    const float* W     = (const float*)d_in[3];   // 512*128
    const float* bias  = (const float*)d_in[4];   // 512
    float* out = (float*)d_out;                   // 200704*512

    const int rows = in_sizes[0] / KDIM;          // 200704
    const int grid = rows / ROWS_PER_BLOCK;       // 1568

    const size_t shmem = (size_t)(ROWS_PER_BLOCK + NOUT) * LDS_STRIDE * sizeof(_Float16); // 174080 B
    (void)hipFuncSetAttribute((const void*)ln_gemm_gelu_kernel,
                              hipFuncAttributeMaxDynamicSharedMemorySize, (int)shmem);

    ln_gemm_gelu_kernel<<<grid, 256, shmem, stream>>>(x, gamma, beta, W, bias, out);
}
